// DynamicDeepHit_78769700208814
// MI455X (gfx1250) — compile-verified, matrix operand order checked by reading, not run on hardware
//
#include <hip/hip_runtime.h>
#include <hip/hip_bf16.h>

typedef __bf16 bf16;
typedef bf16  v16bf __attribute__((ext_vector_type(16)));
typedef bf16  v8bf  __attribute__((ext_vector_type(8)));
typedef float v8f   __attribute__((ext_vector_type(8)));
typedef int   v4i   __attribute__((ext_vector_type(4)));

#define Bn   1024
#define Ln   256
#define Fn   128
#define Hn   256
#define THn  768     // 3*H
#define An   128
#define CHn  256
#define OUTn 512

// ---------------------------------------------------------------------------
// WMMA bf16 fragment loader.
// ISA 16-bit A-matrix 16x32 layout (CDNA5 ISA): lanes 0-15 hold row M=lane,
// K pattern {0..7,16..23}; lanes 16-31 cover {8..15,24..31}: two contiguous
// 16-byte chunks at k0+hs*8 and k0+hs*8+16. B operands use the same pattern
// on the transposed (row-major [N][K]) weight.
// ---------------------------------------------------------------------------
__device__ inline v16bf load_frag(const bf16* p0, const bf16* p1) {
  v8bf lo = *(const v8bf*)p0;
  v8bf hi = *(const v8bf*)p1;
  v16bf r;
#pragma unroll
  for (int i = 0; i < 8; ++i) { r[i] = lo[i]; r[i + 8] = hi[i]; }
  return r;
}

// fragment of a row-major [N][K] weight: lane holds row (n), K-block kb
__device__ inline v16bf fragW(const bf16* W, int row, int ld, int kb, int hs) {
  const bf16* p = W + (size_t)row * ld + kb + hs * 8;
  return load_frag(p, p + 16);
}

__device__ inline v8f wmma_bf16(v16bf a, v16bf b, v8f c) {
  return __builtin_amdgcn_wmma_f32_16x16x32_bf16(false, a, false, b,
                                                 (short)0, c, false, false);
}

__device__ inline float sigmoidf_(float x) {
  return 1.f / (1.f + __expf(-x));
}

// ---------------------------------------------------------------------------
// 16-byte global->LDS copy: async (GLOBAL_LOAD_ASYNC_TO_LDS_B128, ASYNCcnt)
// when the toolchain exposes the builtin, else a plain vector copy.
// Builtin signature (from probe diagnostic): param0 is v4i* (global data).
// ---------------------------------------------------------------------------
#if defined(__has_builtin)
#if __has_builtin(__builtin_amdgcn_global_load_async_to_lds_b128)
#define HAVE_ASYNC_LDS 1
#endif
#endif

__device__ inline void copy16_g2l(const bf16* g, bf16* l) {
#ifdef HAVE_ASYNC_LDS
  __builtin_amdgcn_global_load_async_to_lds_b128(
      (__attribute__((address_space(1))) v4i*)g,
      (__attribute__((address_space(3))) v4i*)l, 0, 0);
#else
  *(v8bf*)l = *(const v8bf*)g;
#endif
}

__device__ inline void wait_async_copies() {
#ifdef HAVE_ASYNC_LDS
#if __has_builtin(__builtin_amdgcn_s_wait_asynccnt)
  __builtin_amdgcn_s_wait_asynccnt(0);
#endif
#endif
}

// ---------------------------------------------------------------------------
// fp32 -> bf16 conversion (input + weights; weights stay hot in 192MB L2)
// ---------------------------------------------------------------------------
__global__ void cvt_bf16_kernel(const float* __restrict__ src,
                                bf16* __restrict__ dst, int n) {
  for (int i = blockIdx.x * blockDim.x + threadIdx.x; i < n;
       i += gridDim.x * blockDim.x)
    dst[i] = (bf16)src[i];
}

// dst[c*R + r] = src[r*C + c]  (transpose Wa so WMMA B reads contiguous K)
__global__ void cvtT_bf16_kernel(const float* __restrict__ src,
                                 bf16* __restrict__ dst, int R, int C) {
  int n = R * C;
  for (int i = blockIdx.x * blockDim.x + threadIdx.x; i < n;
       i += gridDim.x * blockDim.x) {
    int r = i / C, c = i % C;
    dst[c * R + r] = (bf16)src[i];
  }
}

// ---------------------------------------------------------------------------
// Fused GRU recurrence. grid = B/16 workgroups, 256 threads (8 waves).
// Wave w owns H-columns {w*16.., 128+w*16..}; the r/z pre-activations chain
// the x-part and h-part through one accumulator, and the full gate math runs
// in registers on the WMMA accumulators. W_out lives in LDS for the whole
// kernel; x tiles are staged by 16-byte (async) copies of pre-converted bf16;
// the hidden trajectory is written back with vectorized b128 stores.
// ---------------------------------------------------------------------------
__global__ __launch_bounds__(256) void gru_kernel(
    const bf16* __restrict__ xb,    const int*  __restrict__ tte,
    const bf16* __restrict__ Wih,   const bf16* __restrict__ Whh,
    const float* __restrict__ b_ih, const float* __restrict__ b_hh,
    const bf16* __restrict__ Wout,  const float* __restrict__ b_out,
    bf16* __restrict__ hid, float* __restrict__ outb) {
  __shared__ __attribute__((aligned(16))) bf16  x_s[16 * Fn];     //  4 KB
  __shared__ __attribute__((aligned(16))) bf16  h_s[16 * Hn];     //  8 KB frozen h
  __shared__ __attribute__((aligned(16))) bf16  hid_s[16 * Hn];   //  8 KB masked h_new
  __shared__ __attribute__((aligned(16))) bf16  wout_s[Fn * Hn];  // 64 KB W_out
  __shared__ float hf_s[16 * Hn];                                 // 16 KB fp32 h master
  __shared__ int   tte_s[16];

  const int tid  = threadIdx.x;
  const int lane = tid & 31;
  const int wave = tid >> 5;
  const int ln   = lane & 15;        // row (A) / column (B) within tile
  const int hs   = (lane >> 4) & 1;  // lane-half selects K sub-pattern
  const int b0   = blockIdx.x * 16;

  // this lane's gate columns in H-space: c0 = cR in [0,128), c1 = cR+128
  const int cR = wave * 16 + ln;
  const int no = wave * 16;          // this wave's out-projection column tile

  // hoisted per-lane biases (constant over t)
  const float bR0 = b_ih[cR]           + b_hh[cR];
  const float bR1 = b_ih[cR + 128]     + b_hh[cR + 128];
  const float bZ0 = b_ih[cR + 256]     + b_hh[cR + 256];
  const float bZ1 = b_ih[cR + 384]     + b_hh[cR + 384];
  const float bI0 = b_ih[cR + 512];
  const float bI1 = b_ih[cR + 640];
  const float bH0 = b_hh[cR + 512];
  const float bH1 = b_hh[cR + 640];
  const float bo  = b_out[no + ln];

  // x-staging chunk for this thread: 16 rows x 256B = 256 x 16B chunks
  const int xm = tid >> 4;            // row 0..15
  const int xc = (tid & 15) * 8;      // bf16 column 0,8,...,120

  if (tid < 16) tte_s[tid] = tte[b0 + tid];
  for (int i = tid; i < 16 * Hn; i += 256) {
    hf_s[i] = 0.f;
    h_s[i]  = (bf16)0.f;
  }
  // stage W_out (bf16, 128x256) into LDS once; reused all 256 steps
  for (int i = tid; i < Fn * Hn; i += 256) wout_s[i] = Wout[i];
  __syncthreads();

  for (int t = 0; t < Ln; ++t) {
    // ---- stage x_t tile (bf16) into LDS: one b128 copy per thread -------
    copy16_g2l(&xb[((size_t)(b0 + xm) * Ln + t) * Fn + xc], &x_s[xm * Fn + xc]);
    if (t + 1 < Ln && tid < 16)  // global_prefetch_b8 next step's rows
      __builtin_prefetch(&xb[((size_t)(b0 + tid) * Ln + (t + 1)) * Fn], 0, 1);
    wait_async_copies();
    __syncthreads();  // A: x_s ready, previous out-GEMM reads of hid_s done

    // ---- gate pre-activations via WMMA ----------------------------------
    v8f aR0 = {}, aR1 = {}, aZ0 = {}, aZ1 = {};
    v8f aI0 = {}, aI1 = {};  // i_n (x-part only)
    v8f aH0 = {}, aH1 = {};  // h_n (h-part only)
#pragma unroll
    for (int kb = 0; kb < Fn; kb += 32) {  // x @ W_ih^T
      v16bf a = load_frag(&x_s[ln * Fn + kb + hs * 8],
                          &x_s[ln * Fn + kb + hs * 8 + 16]);
      aR0 = wmma_bf16(a, fragW(Wih, cR,       Fn, kb, hs), aR0);
      aR1 = wmma_bf16(a, fragW(Wih, cR + 128, Fn, kb, hs), aR1);
      aZ0 = wmma_bf16(a, fragW(Wih, cR + 256, Fn, kb, hs), aZ0);
      aZ1 = wmma_bf16(a, fragW(Wih, cR + 384, Fn, kb, hs), aZ1);
      aI0 = wmma_bf16(a, fragW(Wih, cR + 512, Fn, kb, hs), aI0);
      aI1 = wmma_bf16(a, fragW(Wih, cR + 640, Fn, kb, hs), aI1);
    }
#pragma unroll
    for (int kb = 0; kb < Hn; kb += 32) {  // h @ W_hh^T (chained into r/z)
      v16bf a = load_frag(&h_s[ln * Hn + kb + hs * 8],
                          &h_s[ln * Hn + kb + hs * 8 + 16]);
      aR0 = wmma_bf16(a, fragW(Whh, cR,       Hn, kb, hs), aR0);
      aR1 = wmma_bf16(a, fragW(Whh, cR + 128, Hn, kb, hs), aR1);
      aZ0 = wmma_bf16(a, fragW(Whh, cR + 256, Hn, kb, hs), aZ0);
      aZ1 = wmma_bf16(a, fragW(Whh, cR + 384, Hn, kb, hs), aZ1);
      aH0 = wmma_bf16(a, fragW(Whh, cR + 512, Hn, kb, hs), aH0);
      aH1 = wmma_bf16(a, fragW(Whh, cR + 640, Hn, kb, hs), aH1);
    }
    __syncthreads();  // B: all reads of h_s complete before gate overwrites

    // ---- gates + state update, in registers; LDS writes only ------------
#pragma unroll
    for (int v = 0; v < 8; ++v) {  // C/D layout: M = v + 8*halfSel, N = ln
      const int m = v + hs * 8;
      const bool act = t < tte_s[m];
      // column pair 0: c = cR
      {
        float r = sigmoidf_(aR0[v] + bR0);
        float z = sigmoidf_(aZ0[v] + bZ0);
        float n = tanhf(aI0[v] + bI0 + r * (aH0[v] + bH0));
        float hold = hf_s[m * Hn + cR];
        float hnew = (1.f - z) * n + z * hold;
        float keep = act ? hnew : hold;
        hf_s[m * Hn + cR]  = keep;
        h_s[m * Hn + cR]   = (bf16)keep;
        hid_s[m * Hn + cR] = (bf16)(act ? hnew : 0.f);
      }
      // column pair 1: c = cR + 128
      {
        float r = sigmoidf_(aR1[v] + bR1);
        float z = sigmoidf_(aZ1[v] + bZ1);
        float n = tanhf(aI1[v] + bI1 + r * (aH1[v] + bH1));
        float hold = hf_s[m * Hn + cR + 128];
        float hnew = (1.f - z) * n + z * hold;
        float keep = act ? hnew : hold;
        hf_s[m * Hn + cR + 128]  = keep;
        h_s[m * Hn + cR + 128]   = (bf16)keep;
        hid_s[m * Hn + cR + 128] = (bf16)(act ? hnew : 0.f);
      }
    }
    __syncthreads();  // C: hid_s complete

    // ---- hidden-trajectory writeback: vectorized b128 (512 x 16B) -------
#pragma unroll
    for (int p = 0; p < 2; ++p) {
      int k = tid + p * 256;          // chunk id; 32 chunks per row
      int m = k >> 5;
      int c = (k & 31) * 8;
      *(v8bf*)&hid[((size_t)(b0 + m) * Ln + t) * Hn + c] =
          *(const v8bf*)&hid_s[m * Hn + c];
    }

    // ---- out_t = hid_s @ W_out^T + b_out (masked); W_out from LDS -------
    {
      v8f ao = {};
#pragma unroll
      for (int k8 = 0; k8 < 8; ++k8) {
        v16bf a = load_frag(&hid_s[ln * Hn + k8 * 32 + hs * 8],
                            &hid_s[ln * Hn + k8 * 32 + hs * 8 + 16]);
        v16bf w = load_frag(&wout_s[(no + ln) * Hn + k8 * 32 + hs * 8],
                            &wout_s[(no + ln) * Hn + k8 * 32 + hs * 8 + 16]);
        ao = wmma_bf16(a, w, ao);
      }
#pragma unroll
      for (int v = 0; v < 8; ++v) {
        int m = v + hs * 8;
        bool act = t < tte_s[m];
        outb[((size_t)(b0 + m) * Ln + t) * Fn + no + ln] =
            act ? (ao[v] + bo) : 0.f;
      }
    }
    __syncthreads();  // D: out-GEMM reads done before next gate phase
  }
}

// ---------------------------------------------------------------------------
// Attention decoder + causes MLP. grid = B workgroups, 256 threads (8 waves).
// T = hid[b] @ Wa via WMMA (16x8 = 128 tiles, 16/wave), rest is GEMV-sized.
// ---------------------------------------------------------------------------
__global__ __launch_bounds__(256) void attn_kernel(
    const float* __restrict__ xin, const int* __restrict__ tte,
    const bf16* __restrict__ hid,  const bf16* __restrict__ WaT,
    const float* __restrict__ Ua,  const float* __restrict__ va,
    const float* __restrict__ W1,  const float* __restrict__ b1,
    const float* __restrict__ W2,  const float* __restrict__ b2,
    float* __restrict__ fout) {
  __shared__ float T_s[Ln * An];   // 128 KB
  __shared__ float last_s[Fn], u_s[An];
  __shared__ float sc_s[Ln], ctx_s[Hn], zc_s[CHn], lg_s[OUTn], red_s[256];

  const int tid  = threadIdx.x;
  const int lane = tid & 31;
  const int wave = tid >> 5;
  const int ln   = lane & 15;
  const int hs   = (lane >> 4) & 1;
  const int b    = blockIdx.x;
  const bf16* hb = hid + (size_t)b * Ln * Hn;

  // last measurement at index tte[b]
  if (tid < Fn)
    last_s[tid] = xin[((size_t)b * Ln + tte[b]) * Fn + tid];
  __syncthreads();

  // u = last @ Ua   [A]
  if (tid < An) {
    float acc = 0.f;
    for (int f = 0; f < Fn; ++f) acc += last_s[f] * Ua[f * An + tid];
    u_s[tid] = acc;
  }
  __syncthreads();

  // T = hid[b] @ Wa  : [L=256, H=256] x [H, A=128] via WMMA bf16
  for (int tIdx = wave; tIdx < 128; tIdx += 8) {
    const int m0 = (tIdx >> 3) * 16;
    const int n0 = (tIdx & 7) * 16;
    v8f acc = {};
#pragma unroll
    for (int kb = 0; kb < Hn; kb += 32) {
      v16bf a = load_frag(&hb[(size_t)(m0 + ln) * Hn + kb + hs * 8],
                          &hb[(size_t)(m0 + ln) * Hn + kb + hs * 8 + 16]);
      v16bf w = fragW(WaT, n0 + ln, Hn, kb, hs);
      acc = wmma_bf16(a, w, acc);
    }
#pragma unroll
    for (int v = 0; v < 8; ++v)
      T_s[(m0 + v + hs * 8) * An + n0 + ln] = acc[v];
  }
  __syncthreads();

  // scores[l] = tanh(T[l,:] + u) . va
  {
    float s = 0.f;
    for (int a = 0; a < An; ++a)
      s += tanhf(T_s[tid * An + a] + u_s[a]) * va[a];
    sc_s[tid] = s;
  }
  __syncthreads();

  // softmax over L=256
  red_s[tid] = sc_s[tid];
  __syncthreads();
  for (int s = 128; s > 0; s >>= 1) {
    if (tid < s) red_s[tid] = fmaxf(red_s[tid], red_s[tid + s]);
    __syncthreads();
  }
  float mx = red_s[0];
  __syncthreads();
  float e = __expf(sc_s[tid] - mx);
  sc_s[tid] = e;
  red_s[tid] = e;
  __syncthreads();
  for (int s = 128; s > 0; s >>= 1) {
    if (tid < s) red_s[tid] += red_s[tid + s];
    __syncthreads();
  }
  float inv = 1.f / red_s[0];
  __syncthreads();
  sc_s[tid] *= inv;  // alpha
  __syncthreads();

  // context[h] = sum_l alpha[l] * hid[b,l,h]
  {
    float c = 0.f;
    for (int l = 0; l < Ln; ++l)
      c += sc_s[l] * (float)hb[(size_t)l * Hn + tid];
    ctx_s[tid] = c;
  }
  __syncthreads();

  // zc = relu([context,last] @ W1 + b1)    W1: [H+F=384][CH=256]
  {
    float acc = b1[tid];
    for (int i = 0; i < Hn; ++i) acc += ctx_s[i] * W1[i * CHn + tid];
    for (int i = 0; i < Fn; ++i) acc += last_s[i] * W1[(Hn + i) * CHn + tid];
    zc_s[tid] = fmaxf(acc, 0.f);
  }
  __syncthreads();

  // logits = zc @ W2 + b2     W2: [CH=256][OUT=512]
#pragma unroll
  for (int p = 0; p < 2; ++p) {
    int o = tid + p * 256;
    float acc = b2[o];
    for (int j = 0; j < CHn; ++j) acc += zc_s[j] * W2[j * OUTn + o];
    lg_s[o] = acc;
  }
  __syncthreads();

  // softmax over OUT=512, write fht
  red_s[tid] = fmaxf(lg_s[tid], lg_s[tid + 256]);
  __syncthreads();
  for (int s = 128; s > 0; s >>= 1) {
    if (tid < s) red_s[tid] = fmaxf(red_s[tid], red_s[tid + s]);
    __syncthreads();
  }
  float m2 = red_s[0];
  __syncthreads();
  float e0 = __expf(lg_s[tid] - m2);
  float e1 = __expf(lg_s[tid + 256] - m2);
  red_s[tid] = e0 + e1;
  __syncthreads();
  for (int s = 128; s > 0; s >>= 1) {
    if (tid < s) red_s[tid] += red_s[tid + s];
    __syncthreads();
  }
  float inv2 = 1.f / red_s[0];
  float* fb = fout + (size_t)b * OUTn;
  fb[tid]       = e0 * inv2;
  fb[tid + 256] = e1 * inv2;
}

// ---------------------------------------------------------------------------
extern "C" void kernel_launch(void* const* d_in, const int* in_sizes, int n_in,
                              void* d_out, int out_size, void* d_ws,
                              size_t ws_size, hipStream_t stream) {
  const float* xin  = (const float*)d_in[0];
  const int*   tte  = (const int*)d_in[1];
  const float* W_ih = (const float*)d_in[2];
  const float* W_hh = (const float*)d_in[3];
  const float* b_ih = (const float*)d_in[4];
  const float* b_hh = (const float*)d_in[5];
  const float* W_out= (const float*)d_in[6];
  const float* b_out= (const float*)d_in[7];
  const float* Wa   = (const float*)d_in[8];
  const float* Ua   = (const float*)d_in[9];
  const float* va   = (const float*)d_in[10];
  const float* W1   = (const float*)d_in[11];
  const float* b1   = (const float*)d_in[12];
  const float* W2   = (const float*)d_in[13];
  const float* b2   = (const float*)d_in[14];

  // workspace layout (bf16)
  char* ws = (char*)d_ws;
  size_t off = 0;
  bf16* hid    = (bf16*)(ws + off); off += (size_t)Bn * Ln * Hn * 2;   // 134 MB
  bf16* xb     = (bf16*)(ws + off); off += (size_t)Bn * Ln * Fn * 2;   //  67 MB
  bf16* Wih_b  = (bf16*)(ws + off); off += (size_t)THn * Fn * 2;
  bf16* Whh_b  = (bf16*)(ws + off); off += (size_t)THn * Hn * 2;
  bf16* Wout_b = (bf16*)(ws + off); off += (size_t)Fn * Hn * 2;
  bf16* WaT_b  = (bf16*)(ws + off); off += (size_t)An * Hn * 2;

  float* outb = (float*)d_out;                       // [B,L,F]
  float* fout = outb + (size_t)Bn * Ln * Fn;         // [B,OUT]

  cvt_bf16_kernel<<<8192, 256, 0, stream>>>(xin, xb, Bn * Ln * Fn);
  cvt_bf16_kernel<<<(THn * Fn + 255) / 256, 256, 0, stream>>>(W_ih, Wih_b, THn * Fn);
  cvt_bf16_kernel<<<(THn * Hn + 255) / 256, 256, 0, stream>>>(W_hh, Whh_b, THn * Hn);
  cvt_bf16_kernel<<<(Fn * Hn + 255) / 256, 256, 0, stream>>>(W_out, Wout_b, Fn * Hn);
  cvtT_bf16_kernel<<<(Hn * An + 255) / 256, 256, 0, stream>>>(Wa, WaT_b, Hn, An);

  gru_kernel<<<Bn / 16, 256, 0, stream>>>(xb, tte, Wih_b, Whh_b, b_ih, b_hh,
                                          Wout_b, b_out, hid, outb);

  attn_kernel<<<Bn, 256, 0, stream>>>(xin, tte, hid, WaT_b, Ua, va, W1, b1,
                                      W2, b2, fout);
}
